// DifferentialAttention_81750407512335
// MI455X (gfx1250) — matrix operand order repaired and checked by reading, not compiled, not run on hardware
//
#include <hip/hip_runtime.h>
#include <hip/hip_bf16.h>

typedef _Float16 f16;
typedef __attribute__((ext_vector_type(16))) _Float16 v16h;
typedef __attribute__((ext_vector_type(8)))  float    v8f;

#define S_LEN  4096
#define DMODEL 1024
#define ROTD   128
#define NB     8      // ROTD / 16
#define PAD    8      // 16-byte row pad (f16 units) to dodge bank conflicts

__device__ __forceinline__ int lane_id() { return threadIdx.x & 31; }

// Keep fragment loads batched ahead of the WMMA chain: nothing may cross.
__device__ __forceinline__ void sched_fence() { __builtin_amdgcn_sched_barrier(0); }

__device__ __forceinline__ v8f wmma16(v16h a, v16h b, v8f c) {
  // D = A(16x32 f16) * B(32x16 f16) + C(16x16 f32)
  return __builtin_amdgcn_wmma_f32_16x16x32_f16(false, a, false, b, (short)0, c,
                                                false, false);
}

// A fragment: 16x32 f16 tile, row-major, leading dim ld (halves).
// ISA 16-bit A layout: lanes 0-15 row M=l, K={kb..kb+7, 16+kb..16+kb+7}, kb=0;
// lanes 16-31 same rows, kb=8.  Per lane: two contiguous 16B chunks.
__device__ __forceinline__ v16h load_a_frag(const f16* src, int ld) {
  const int l = lane_id();
  const int m = l & 15;
  const int kb = (l < 16) ? 0 : 8;
  v16h a;
#pragma unroll
  for (int e = 0; e < 8; ++e) a[e] = src[m * ld + kb + e];
#pragma unroll
  for (int e = 0; e < 8; ++e) a[8 + e] = src[m * ld + 16 + kb + e];
  return a;
}

// B fragment (K=32 x N=16) from N-major storage: B[k][n] = src[n*ld + k].
// Per lane: K={kb..kb+7, 16+kb..16+kb+7}, kb=0/8 -> two 16B chunks.
__device__ __forceinline__ v16h load_bT_frag(const f16* src, int ld) {
  const int l = lane_id();
  const int n = l & 15;
  const int kb = (l < 16) ? 0 : 8;
  v16h b;
#pragma unroll
  for (int e = 0; e < 8; ++e) b[e] = src[n * ld + kb + e];
#pragma unroll
  for (int e = 0; e < 8; ++e) b[8 + e] = src[n * ld + 16 + kb + e];
  return b;
}

__device__ __forceinline__ float rowmax16(float v) {
#pragma unroll
  for (int m = 8; m >= 1; m >>= 1) v = fmaxf(v, __shfl_xor(v, m, 16));
  return v;
}
__device__ __forceinline__ float rowsum16(float v) {
#pragma unroll
  for (int m = 8; m >= 1; m >>= 1) v += __shfl_xor(v, m, 16);
  return v;
}

// Dual-fragment online-softmax update. sa/sb are the two 16-key score tiles
// (D layout); on return they hold exp(s - m_new). O (8 N-blocks) is rescaled.
__device__ __forceinline__ void online_update(v8f& sa, v8f& sb,
                                              float* m, float* s, v8f* O) {
#pragma unroll
  for (int r = 0; r < 8; ++r) {
    float loc = rowmax16(fmaxf(sa[r], sb[r]));
    float mn = fmaxf(m[r], loc);
    float scale = __expf(m[r] - mn);
    m[r] = mn;
    float pa = __expf(sa[r] - mn);
    float pb = __expf(sb[r] - mn);
    sa[r] = pa;
    sb[r] = pb;
    s[r] = s[r] * scale + rowsum16(pa + pb);
#pragma unroll
    for (int nb = 0; nb < NB; ++nb) O[nb][r] *= scale;
  }
}

// ---------------------------------------------------------------------------
// lambda = exp(sum(lq1*lk1)) - exp(sum(lq2*lk2)) + lambda_init
__global__ void lambda_kernel(const float* __restrict__ lq1,
                              const float* __restrict__ lq2,
                              const float* __restrict__ lk1,
                              const float* __restrict__ lk2,
                              float* __restrict__ lam) {
  if (threadIdx.x == 0 && blockIdx.x == 0) {
    float a = 0.f, b = 0.f;
    for (int i = 0; i < 64; ++i) {
      a += lq1[i] * lk1[i];
      b += lq2[i] * lk2[i];
    }
    float lambda_init = 0.8f - 0.6f * __expf(-0.3f);
    *lam = __expf(a) - __expf(b) + lambda_init;
  }
}

// ---------------------------------------------------------------------------
// out[row,0..127] = rope?( x[row,:] @ W ) * out_scale, stored as f16.
// Block: 128 threads (4 waves); each wave owns 16 rows; block owns 64 rows.
// W is staged TRANSPOSED in LDS so B fragments are contiguous per lane.
__global__ __launch_bounds__(128)
void proj_rope_kernel(const float* __restrict__ x, const float* __restrict__ W,
                      f16* __restrict__ out, int do_rope, float out_scale) {
  __shared__ __align__(16) f16 ldsX[64][32 + PAD];
  __shared__ __align__(16) f16 ldsWt[ROTD][32 + PAD];   // [dim][k]
  const int t = threadIdx.x;
  const int wave = t >> 5;
  const int l = t & 31;
  const int rowbase = blockIdx.x * 64;

  v8f acc[NB] = {};

  for (int kc = 0; kc < DMODEL; kc += 32) {
    // stage x tile 64x32 (f32 -> f16), row-major padded
#pragma unroll
    for (int i = 0; i < 16; ++i) {
      int idx = t * 16 + i;                 // 0..2047
      int r = idx >> 5, c = idx & 31;
      ldsX[r][c] = (f16)x[(size_t)(rowbase + r) * DMODEL + kc + c];
    }
    // stage W tile 32x128 (f32 -> f16), transposed: ldsWt[dim][k]
#pragma unroll
    for (int i = 0; i < 32; ++i) {
      int idx = t * 32 + i;                 // 0..4095
      int r = idx >> 7, c = idx & 127;      // r = k (0..31), c = dim (0..127)
      ldsWt[c][r] = (f16)W[(size_t)(kc + r) * ROTD + c];
    }
    __syncthreads();
    // batch ALL fragment loads (kept above the fence), then the WMMA chain:
    // forces graduated s_wait_dscnt instead of a full drain per WMMA.
    v16h a = load_a_frag(&ldsX[wave * 16][0], 32 + PAD);
    v16h bfr[NB];
#pragma unroll
    for (int nb = 0; nb < NB; ++nb)
      bfr[nb] = load_bT_frag(&ldsWt[nb * 16][0], 32 + PAD);
    sched_fence();
#pragma unroll
    for (int nb = 0; nb < NB; ++nb) acc[nb] = wmma16(a, bfr[nb], acc[nb]);
    __syncthreads();
  }

  // Epilogue: RoPE (lane-local in D layout) + f16 store.
  const int n = l & 15;
  const int roff = (l < 16) ? 0 : 8;
  const float kfreq = -9.2103403719761836f / 64.0f;  // -ln(10000)/64
#pragma unroll
  for (int r = 0; r < 8; ++r) {
    int row = rowbase + wave * 16 + r + roff;
    float srow = (float)(row & (S_LEN - 1));
    float val[NB];
    if (do_rope) {
#pragma unroll
      for (int nb = 0; nb < NB; ++nb) {
        int i = nb * 16 + n;
        int j = i & 63;
        float ang = srow * __expf(kfreq * (float)j);
        float sv, cv;
        __sincosf(ang, &sv, &cv);
        float rot = (nb < 4) ? -acc[nb + 4][r] : acc[nb - 4][r];
        val[nb] = acc[nb][r] * cv + rot * sv;
      }
    } else {
#pragma unroll
      for (int nb = 0; nb < NB; ++nb) val[nb] = acc[nb][r];
    }
#pragma unroll
    for (int nb = 0; nb < NB; ++nb)
      out[(size_t)row * ROTD + nb * 16 + n] = (f16)(val[nb] * out_scale);
  }
}

// ---------------------------------------------------------------------------
// Fused differential flash attention.
// grid (S/64, B), block 128; each wave owns a 16-row query tile; all 4 waves
// share K (row-major padded) and V (transposed [dim][key]) tiles in LDS.
__global__ __launch_bounds__(128)
void diff_attn_kernel(const f16* __restrict__ Qh, const f16* __restrict__ Kh,
                      const f16* __restrict__ Vh, const float* __restrict__ lam_p,
                      float* __restrict__ out) {
  __shared__ __align__(16) f16 ldsK[32][ROTD + PAD];       // [key][dim]
  __shared__ __align__(16) f16 ldsVt[ROTD][32 + PAD];      // [dim][key]
  __shared__ __align__(16) f16 ldsP[4][2][16][32 + PAD];   // per-wave, x2 buf

  const int t = threadIdx.x;
  const int wave = t >> 5;
  const int l = t & 31;
  const int n = l & 15;
  const int roff = (l < 16) ? 0 : 8;
  const int b = blockIdx.y;
  const int q0 = blockIdx.x * 64 + wave * 16;
  const size_t gq = (size_t)b * S_LEN + q0;

  const float lam = *lam_p;

  // Q fragments (1/sqrt(HD) already folded into Qh)
  const f16* qp = Qh + gq * ROTD;
  v16h q1c0 = load_a_frag(qp + 0, ROTD);
  v16h q1c1 = load_a_frag(qp + 32, ROTD);
  v16h q2c0 = load_a_frag(qp + 64, ROTD);
  v16h q2c1 = load_a_frag(qp + 96, ROTD);

  v8f O1[NB] = {}, O2[NB] = {};
  float m1[8], s1[8], m2[8], s2[8];
#pragma unroll
  for (int r = 0; r < 8; ++r) { m1[r] = m2[r] = -1e30f; s1[r] = s2[r] = 0.f; }

  for (int kt = 0; kt < S_LEN; kt += 32) {
    const f16* Ksrc = Kh + ((size_t)b * S_LEN + kt) * ROTD;
    const f16* Vsrc = Vh + ((size_t)b * S_LEN + kt) * ROTD;
    // K: straight copy, 16B vectors into padded rows. 512 uint4 / 128 thr.
#pragma unroll
    for (int i = 0; i < 4; ++i) {
      int j = t + i * 128;                  // 0..511
      int key = j >> 4, seg = j & 15;       // 16 uint4 per 128-half row
      *(uint4*)&ldsK[key][seg * 8] =
          *(const uint4*)(Ksrc + (size_t)key * ROTD + seg * 8);
    }
    // V: vector read, transposed scatter into ldsVt[dim][key].
#pragma unroll
    for (int i = 0; i < 4; ++i) {
      int j = t + i * 128;
      int key = j >> 4, seg = j & 15;
      uint4 raw = *(const uint4*)(Vsrc + (size_t)key * ROTD + seg * 8);
      const f16* vv = (const f16*)&raw;
#pragma unroll
      for (int u = 0; u < 8; ++u) ldsVt[seg * 8 + u][key] = vv[u];
    }
    __syncthreads();

    // scores: batch all 8 K B-fragments, fence, then 8 WMMAs
    v16h kf[8];
#pragma unroll
    for (int c = 0; c < 4; ++c) {
      kf[c]     = load_bT_frag(&ldsK[0][c * 32],  ROTD + PAD);
      kf[4 + c] = load_bT_frag(&ldsK[16][c * 32], ROTD + PAD);
    }
    sched_fence();
    v8f sA1 = {}, sB1 = {}, sA2 = {}, sB2 = {};
    sA1 = wmma16(q1c0, kf[0], sA1);
    sB1 = wmma16(q1c0, kf[4], sB1);
    sA2 = wmma16(q2c0, kf[2], sA2);
    sB2 = wmma16(q2c0, kf[6], sB2);
    sA1 = wmma16(q1c1, kf[1], sA1);
    sB1 = wmma16(q1c1, kf[5], sB1);
    sA2 = wmma16(q2c1, kf[3], sA2);
    sB2 = wmma16(q2c1, kf[7], sB2);

    // dual online softmax
    online_update(sA1, sB1, m1, s1, O1);
    online_update(sA2, sB2, m2, s2, O2);

    // re-layout P (D-layout f32 -> A-layout f16) through per-wave LDS
    // scratch; two buffers so both writes drain before both reads
    f16(*pP1)[32 + PAD] = ldsP[wave][0];
    f16(*pP2)[32 + PAD] = ldsP[wave][1];
#pragma unroll
    for (int r = 0; r < 8; ++r) {
      pP1[r + roff][n]      = (f16)sA1[r];
      pP1[r + roff][n + 16] = (f16)sB1[r];
      pP2[r + roff][n]      = (f16)sA2[r];
      pP2[r + roff][n + 16] = (f16)sB2[r];
    }
    v16h p1 = load_a_frag(&pP1[0][0], 32 + PAD);
    v16h p2 = load_a_frag(&pP2[0][0], 32 + PAD);

    // O += P * V : batch V fragments, fence, share each between both accs
    v16h vf[NB];
#pragma unroll
    for (int nb = 0; nb < NB; ++nb)
      vf[nb] = load_bT_frag(&ldsVt[nb * 16][0], 32 + PAD);
    sched_fence();
#pragma unroll
    for (int nb = 0; nb < NB; ++nb) {
      O1[nb] = wmma16(p1, vf[nb], O1[nb]);
      O2[nb] = wmma16(p2, vf[nb], O2[nb]);
    }
    __syncthreads();
  }

  // out = O1/s1 - lambda * O2/s2
#pragma unroll
  for (int r = 0; r < 8; ++r) {
    float inv1 = 1.0f / s1[r];
    float inv2 = lam / s2[r];
    size_t row = gq + r + roff;
#pragma unroll
    for (int nb = 0; nb < NB; ++nb)
      out[row * ROTD + nb * 16 + n] = O1[nb][r] * inv1 - O2[nb][r] * inv2;
  }
}

// ---------------------------------------------------------------------------
extern "C" void kernel_launch(void* const* d_in, const int* in_sizes, int n_in,
                              void* d_out, int out_size, void* d_ws, size_t ws_size,
                              hipStream_t stream) {
  const float* x   = (const float*)d_in[0];
  const float* Wq  = (const float*)d_in[1];
  const float* Wk  = (const float*)d_in[2];
  const float* Wv  = (const float*)d_in[3];
  const float* lq1 = (const float*)d_in[4];
  const float* lq2 = (const float*)d_in[5];
  const float* lk1 = (const float*)d_in[6];
  const float* lk2 = (const float*)d_in[7];
  float* out = (float*)d_out;

  const size_t rows = 2 * (size_t)S_LEN;  // B*S = 8192
  f16* Qh = (f16*)d_ws;
  f16* Kh = Qh + rows * ROTD;
  f16* Vh = Kh + rows * ROTD;
  float* lamp = (float*)(Vh + rows * ROTD);

  lambda_kernel<<<1, 32, 0, stream>>>(lq1, lq2, lk1, lk2, lamp);
  // fold score scale 1/sqrt(64)=0.125 into stored Q
  proj_rope_kernel<<<rows / 64, 128, 0, stream>>>(x, Wq, Qh, 1, 0.125f);
  proj_rope_kernel<<<rows / 64, 128, 0, stream>>>(x, Wk, Kh, 1, 1.0f);
  proj_rope_kernel<<<rows / 64, 128, 0, stream>>>(x, Wv, Vh, 0, 1.0f);
  diff_attn_kernel<<<dim3(S_LEN / 64, 2), 128, 0, stream>>>(Qh, Kh, Vh, lamp, out);
}
